// ProteinGATv2Encoder_12068858101899
// MI455X (gfx1250) — matrix operand order repaired and checked by reading, not compiled, not run on hardware
//
#include <hip/hip_runtime.h>
#include <hip/hip_bf16.h>
#include <math.h>

// ============================================================================
// ProteinGATv2Encoder for MI455X (gfx1250, wave32, WMMA).
//
//  * All dense GEMMs run through v_wmma_f32_16x16x32_f16. One wave computes a
//    16 x (16*NT) strip (NT=8 for N>=128, NT=4 for N=64): NT accumulators
//    share one A fragment -> NT back-to-back WMMAs per k-step and A-operand
//    reuse NT-fold. For the dominant M=800k ee GEMM (N=128) the A stream is
//    read exactly once.
//  * Operands are pre-packed f16, K padded to a multiple of 32: A loads are
//    two contiguous b128s per fragment, B is packed [K/32][N][32] so each
//    lane's 16 values are one contiguous 32B load. No bounds checks, no
//    divergent scalar loads in the inner loop.
//  * ee [E,128] (the dominant HBM stream) is produced and consumed in f16:
//    205 MB instead of 410 MB -> halves the roofline term at 23.3 TB/s.
//  * Segment softmax/scatter use float atomics on node-sized buffers
//    (25.6 MB) that live in the 192 MB L2.
// ============================================================================

typedef __attribute__((ext_vector_type(16))) _Float16 v16h;
typedef __attribute__((ext_vector_type(8)))  _Float16 v8h;
typedef __attribute__((ext_vector_type(8)))  float    v8f;

__device__ __forceinline__ float gelu_erf(float x) {
  return 0.5f * x * (1.0f + erff(x * 0.70710678118654752f));
}

// Order-preserving float -> uint key so atomicMax(uint) == float max.
__device__ __forceinline__ unsigned fkey(float v) {
  unsigned u = __float_as_uint(v);
  return (u & 0x80000000u) ? ~u : (u | 0x80000000u);
}
__device__ __forceinline__ float fkey_dec(unsigned k) {
  unsigned u = (k & 0x80000000u) ? (k & 0x7fffffffu) : ~k;
  return __uint_as_float(u);
}
#define FKEY_NEG_INF 0x007FFFFFu  // fkey(-inf)

__device__ __forceinline__ float bn10(float v, float lo, float hi) {
  float vn = (fminf(fmaxf(v, lo), hi) - lo) / (hi - lo);
  return floorf(vn * 10.0f) * 0.1f;
}

// ---------------------------------------------------------------------------
// Utility fills / weight packing
// ---------------------------------------------------------------------------
__global__ void fill_f32(float* p, float v, long n) {
  long i = (long)blockIdx.x * blockDim.x + threadIdx.x;
  if (i < n) p[i] = v;
}
__global__ void fill_u32(unsigned* p, unsigned v, long n) {
  long i = (long)blockIdx.x * blockDim.x + threadIdx.x;
  if (i < n) p[i] = v;
}

// Pack f32 row-major B[K,N] into f16 tiles Bp[(K32/32)][N][32], zero-padding
// rows K..K32-1. One-time, tiny.
__global__ void pack_b(const float* __restrict__ B, _Float16* __restrict__ Bp,
                       int N, int K, int K32) {
  long i = (long)blockIdx.x * blockDim.x + threadIdx.x;
  long total = (long)K32 * N;
  if (i >= total) return;
  int  kk = (int)(i & 31);
  long t  = i >> 5;
  int  n  = (int)(t % N);
  int  kb = (int)(t / N);
  int  k  = kb * 32 + kk;
  Bp[i] = (_Float16)((k < K) ? B[(size_t)k * N + n] : 0.0f);
}

// ---------------------------------------------------------------------------
// WMMA GEMM: C[M,N] = act(A[M,K] @ B[K,N] + bias)
//   A : f16 row-major [M,K], K a multiple of 32 (caller pads).
//   Bp: f16 packed [(K/32)][N][32].
//   Cf (f32) and Ch (f16) are optional outputs (either may be null).
// One wave computes a 16 x (16*NT) strip: NT accumulators share one A
// fragment. N must be a multiple of 16*NT, M a multiple of 16.
// Tile guard is wave-uniform so EXEC is all-ones at the WMMAs.
// A and B fragments use contiguous k = 16*half + s (this code's order); C layout per CDNA5 ISA 7.12.2.
// ---------------------------------------------------------------------------
template <int ACT, int NT>
__global__ void gemm_wmma(const _Float16* __restrict__ A, const _Float16* __restrict__ Bp,
                          const float* __restrict__ bias,
                          float* __restrict__ Cf, _Float16* __restrict__ Ch,
                          int M, int N, int K) {
  const int lane = threadIdx.x & 31;
  const int wv   = threadIdx.x >> 5;
  const int half = lane >> 4;
  const int l15  = lane & 15;
  const int tiles_m = M >> 4;
  const int grp_n   = N / (16 * NT);  // strip groups along N
  int t = blockIdx.x * (blockDim.x >> 5) + wv;
  if (t >= tiles_m * grp_n) return;  // wave-uniform
  const int tm = t / grp_n;
  const int tg = t - tm * grp_n;
  const int m  = tm * 16 + l15;      // A row held by this lane
  const int nb = tg * (16 * NT);     // first column of this wave's strip

  v8f cacc[NT];
#pragma unroll
  for (int j = 0; j < NT; ++j) cacc[j] = (v8f){};

  const _Float16* arow = A + (size_t)m * K + 16 * half;
  for (int k0 = 0; k0 < K; k0 += 32) {
    v8h alo = *(const v8h*)(arow + k0);        // K = k0+16h .. +7
    v8h ahi = *(const v8h*)(arow + k0 + 8);    // K = k0+16h+8 .. +7
    v16h a = __builtin_shufflevector(alo, ahi, 0, 1, 2, 3, 4, 5, 6, 7,
                                     8, 9, 10, 11, 12, 13, 14, 15);
    const _Float16* bb =
        Bp + ((size_t)(k0 >> 5) * N + nb + l15) * 32 + 16 * half;
#pragma unroll
    for (int j = 0; j < NT; ++j) {
      v16h b = *(const v16h*)(bb + (size_t)j * 16 * 32);
      cacc[j] = __builtin_amdgcn_wmma_f32_16x16x32_f16(false, a, false, b,
                                                       (short)0, cacc[j], false, false);
    }
  }
#pragma unroll
  for (int j = 0; j < NT; ++j) {
    int col = nb + 16 * j + l15;
    float bv = bias ? bias[col] : 0.0f;
#pragma unroll
    for (int i = 0; i < 8; ++i) {
      int row = tm * 16 + i + 8 * half;
      float v = cacc[j][i] + bv;
      if (ACT == 1) v = gelu_erf(v);
      if (Cf) Cf[(size_t)row * N + col] = v;
      if (Ch) Ch[(size_t)row * N + col] = (_Float16)v;
    }
  }
}

// ---------------------------------------------------------------------------
// Position encoder: per-edge dir -> MLP(3->32, LN, GELU, 32->64), scatter-sum
// ---------------------------------------------------------------------------
__global__ void pos_edge_kernel(const float* __restrict__ pos,
                                const int* __restrict__ src, const int* __restrict__ dst,
                                const float* __restrict__ w1, const float* __restrict__ b1,
                                const float* __restrict__ g1, const float* __restrict__ be1,
                                const float* __restrict__ w2, const float* __restrict__ b2,
                                float* __restrict__ acc, float* __restrict__ deg, long E) {
  long e = (long)blockIdx.x * blockDim.x + threadIdx.x;
  if (e >= E) return;
  int s0 = src[e], d0 = dst[e];
  float r0 = pos[(size_t)d0 * 3 + 0] - pos[(size_t)s0 * 3 + 0];
  float r1 = pos[(size_t)d0 * 3 + 1] - pos[(size_t)s0 * 3 + 1];
  float r2 = pos[(size_t)d0 * 3 + 2] - pos[(size_t)s0 * 3 + 2];
  float d = sqrtf(r0 * r0 + r1 * r1 + r2 * r2) + 1e-6f;
  r0 /= d; r1 /= d; r2 /= d;
  float hb[32];
  float mean = 0.0f;
#pragma unroll
  for (int j = 0; j < 32; ++j) {
    float s = b1[j] + r0 * w1[j] + r1 * w1[32 + j] + r2 * w1[64 + j];
    hb[j] = s; mean += s;
  }
  mean *= (1.0f / 32.0f);
  float var = 0.0f;
#pragma unroll
  for (int j = 0; j < 32; ++j) { float tt = hb[j] - mean; var += tt * tt; }
  var *= (1.0f / 32.0f);
  float inv = rsqrtf(var + 1e-5f);
#pragma unroll
  for (int j = 0; j < 32; ++j)
    hb[j] = gelu_erf((hb[j] - mean) * inv * g1[j] + be1[j]);
  for (int j = 0; j < 64; ++j) {
    float s = b2[j];
#pragma unroll
    for (int i = 0; i < 32; ++i) s += hb[i] * w2[i * 64 + j];
    atomicAdd(&acc[(size_t)d0 * 64 + j], s);
  }
  atomicAdd(&deg[d0], 1.0f);
}

// xin_h[N,96] f16 = [property-normalized x (24), pe = acc/deg (64), pad (8)]
__global__ void build_xin(const float* __restrict__ x, const float* __restrict__ acc,
                          const float* __restrict__ deg, _Float16* __restrict__ xin,
                          int N) {
  int n = blockIdx.x * blockDim.x + threadIdx.x;
  if (n >= N) return;
  _Float16* p = xin + (size_t)n * 96;
  for (int c = 0; c < 24; ++c) {
    float v = x[(size_t)n * 24 + c];
    if (c == 0) v = bn10(v, -4.5f, 4.5f);
    else if (c == 1) v = bn10(v, -1.0f, 1.0f);
    else if (c == 3) v = bn10(v, 75.0f, 204.0f);
    p[c] = (_Float16)v;
  }
  float invd = 1.0f / fmaxf(deg[n], 1.0f);
  for (int j = 0; j < 64; ++j)
    p[24 + j] = (_Float16)(acc[(size_t)n * 64 + j] * invd);
  for (int c = 88; c < 96; ++c) p[c] = (_Float16)0.0f;
}

// ein_h[E,32] f16 = [edge_attr(16), one_hot(edge_type,2), pad(14)]
__global__ void build_ein(const float* __restrict__ ea, const int* __restrict__ et,
                          _Float16* __restrict__ ein, long E) {
  long e = (long)blockIdx.x * blockDim.x + threadIdx.x;
  if (e >= E) return;
  _Float16* p = ein + (size_t)e * 32;
  for (int c = 0; c < 16; ++c) p[c] = (_Float16)ea[(size_t)e * 16 + c];
  int t = et[e];
  p[16] = (_Float16)((t == 0) ? 1.0f : 0.0f);
  p[17] = (_Float16)((t == 1) ? 1.0f : 0.0f);
  for (int c = 18; c < 32; ++c) p[c] = (_Float16)0.0f;
}

// LayerNorm + GELU over rows of width D: read f32, write f16.
__global__ void ln_gelu_rows_f16(const float* __restrict__ X, _Float16* __restrict__ O,
                                 const float* __restrict__ g, const float* __restrict__ b,
                                 long rows, int D) {
  long r = (long)blockIdx.x * blockDim.x + threadIdx.x;
  if (r >= rows) return;
  const float* p = X + (size_t)r * D;
  float mean = 0.0f;
  for (int c = 0; c < D; ++c) mean += p[c];
  mean /= (float)D;
  float var = 0.0f;
  for (int c = 0; c < D; ++c) { float t = p[c] - mean; var += t * t; }
  var /= (float)D;
  float inv = rsqrtf(var + 1e-5f);
  _Float16* o = O + (size_t)r * D;
  for (int c = 0; c < D; ++c)
    o[c] = (_Float16)gelu_erf((p[c] - mean) * inv * g[c] + b[c]);
}

// ---------------------------------------------------------------------------
// GATv2 attention passes (H*dh == 128 for every layer); ee is f16.
// ---------------------------------------------------------------------------
__global__ void att_logits_max(const float* __restrict__ xl, const float* __restrict__ xr,
                               const _Float16* __restrict__ ee,
                               const int* __restrict__ src, const int* __restrict__ dst,
                               const float* __restrict__ att,
                               float* __restrict__ logit, unsigned* __restrict__ mkey,
                               long E, int H, int dh) {
  long e = (long)blockIdx.x * blockDim.x + threadIdx.x;
  if (e >= E) return;
  int s0 = src[e], d0 = dst[e];
  const float* pl = xl + (size_t)s0 * 128;
  const float* pr = xr + (size_t)d0 * 128;
  const _Float16* pe = ee + (size_t)e * 128;
  for (int h = 0; h < H; ++h) {
    const float* ah = att + h * dh;
    int base = h * dh;
    float acc = 0.0f;
    for (int d = 0; d < dh; ++d) {
      float sv = pl[base + d] + pr[base + d] + (float)pe[base + d];
      sv = sv > 0.0f ? sv : 0.2f * sv;  // leaky_relu(0.2)
      acc += ah[d] * sv;
    }
    logit[e * H + h] = acc;
    atomicMax(&mkey[(size_t)d0 * H + h], fkey(acc));
  }
}

__global__ void att_expsum(const float* __restrict__ logit, const int* __restrict__ dst,
                           const unsigned* __restrict__ mkey, float* __restrict__ ssum,
                           long E, int H) {
  long e = (long)blockIdx.x * blockDim.x + threadIdx.x;
  if (e >= E) return;
  int d0 = dst[e];
  for (int h = 0; h < H; ++h) {
    float m = fkey_dec(mkey[(size_t)d0 * H + h]);
    if (!(m > -1e37f)) m = 0.0f;  // where(isfinite(m), m, 0)
    atomicAdd(&ssum[(size_t)d0 * H + h], expf(logit[e * H + h] - m));
  }
}

// 4 threads per edge, 32 channels each; scatter msg = xl[src]*alpha onto dst.
__global__ void att_scatter(const float* __restrict__ xl, const float* __restrict__ logit,
                            const int* __restrict__ src, const int* __restrict__ dst,
                            const unsigned* __restrict__ mkey, const float* __restrict__ ssum,
                            float* __restrict__ aggr, long E, int H, int dh) {
  long idx = (long)blockIdx.x * blockDim.x + threadIdx.x;
  if (idx >= E * 4) return;
  long e = idx >> 2;
  int q = (int)(idx & 3);
  int s0 = src[e], d0 = dst[e];
  int c0 = q * 32;
  int hPrev = -1;
  float alpha = 0.0f;
  for (int j = 0; j < 32; ++j) {
    int c = c0 + j;
    int h = c / dh;
    if (h != hPrev) {
      float m = fkey_dec(mkey[(size_t)d0 * H + h]);
      if (!(m > -1e37f)) m = 0.0f;
      float den = ssum[(size_t)d0 * H + h] + 1e-16f;
      alpha = expf(logit[e * H + h] - m) / den;
      hPrev = h;
    }
    atomicAdd(&aggr[(size_t)d0 * 128 + c], xl[(size_t)s0 * 128 + c] * alpha);
  }
}

// out = gelu(LN(aggr + h_old)); written to h_new (f32+f16) and concat (f16).
__global__ void residual_ln_gelu(const float* __restrict__ aggr, const float* __restrict__ hold,
                                 const float* __restrict__ g, const float* __restrict__ b,
                                 float* __restrict__ hnew, _Float16* __restrict__ hnewh,
                                 _Float16* __restrict__ cath, int catOff, int N) {
  int n = blockIdx.x * blockDim.x + threadIdx.x;
  if (n >= N) return;
  const float* pa = aggr + (size_t)n * 128;
  const float* ph = hold + (size_t)n * 128;
  float mean = 0.0f;
  for (int c = 0; c < 128; ++c) mean += pa[c] + ph[c];
  mean *= (1.0f / 128.0f);
  float var = 0.0f;
  for (int c = 0; c < 128; ++c) { float t = pa[c] + ph[c] - mean; var += t * t; }
  var *= (1.0f / 128.0f);
  float inv = rsqrtf(var + 1e-5f);
  float*    pn = hnew  + (size_t)n * 128;
  _Float16* pH = hnewh + (size_t)n * 128;
  _Float16* pc = cath  + (size_t)n * 384 + catOff;
  for (int c = 0; c < 128; ++c) {
    float v = gelu_erf((pa[c] + ph[c] - mean) * inv * g[c] + b[c]);
    pn[c] = v;
    pH[c] = (_Float16)v;
    pc[c] = (_Float16)v;
  }
}

// Final importance head matvec: scores = hid[N,128] @ w2[128,1] + b2
__global__ void imp_final(const float* __restrict__ hid, const float* __restrict__ w2,
                          const float* __restrict__ b2, float* __restrict__ scores, int N) {
  int n = blockIdx.x * blockDim.x + threadIdx.x;
  if (n >= N) return;
  float s = b2[0];
  const float* p = hid + (size_t)n * 128;
  for (int c = 0; c < 128; ++c) s += p[c] * w2[c];
  scores[n] = s;
}

// ---------------------------------------------------------------------------
// Host orchestration
// ---------------------------------------------------------------------------
static inline int cdiv(long a, long b) { return (int)((a + b - 1) / b); }

extern "C" void kernel_launch(void* const* d_in, const int* in_sizes, int n_in,
                              void* d_out, int out_size, void* d_ws, size_t ws_size,
                              hipStream_t stream) {
  (void)n_in; (void)out_size; (void)ws_size;
  // Inputs in setup_inputs() recursive dict-insertion order:
  const float* x    = (const float*)d_in[0];
  const int*   eidx = (const int*)d_in[1];
  const float* ea   = (const float*)d_in[2];
  const int*   et   = (const int*)d_in[3];
  const float* pos  = (const float*)d_in[4];
  // d_in[5] = batch (unused)
  const float* p_w1  = (const float*)d_in[6];
  const float* p_b1  = (const float*)d_in[7];
  const float* p_g1  = (const float*)d_in[8];
  const float* p_be1 = (const float*)d_in[9];
  const float* p_w2  = (const float*)d_in[10];
  const float* p_b2  = (const float*)d_in[11];
  const float* n_w1  = (const float*)d_in[12];
  const float* n_b1  = (const float*)d_in[13];
  const float* n_w2  = (const float*)d_in[14];
  const float* n_b2  = (const float*)d_in[15];
  const float* e_w   = (const float*)d_in[16];
  const float* e_b   = (const float*)d_in[17];
  const float* e_g   = (const float*)d_in[18];
  const float* e_be  = (const float*)d_in[19];
  const float* f_w1  = (const float*)d_in[38];
  const float* f_b1  = (const float*)d_in[39];
  const float* f_g   = (const float*)d_in[40];
  const float* f_be  = (const float*)d_in[41];
  const float* f_w2  = (const float*)d_in[42];
  const float* f_b2  = (const float*)d_in[43];
  const float* i_w1  = (const float*)d_in[44];
  const float* i_b1  = (const float*)d_in[45];
  const float* i_w2  = (const float*)d_in[46];
  const float* i_b2  = (const float*)d_in[47];

  const int  N = in_sizes[0] / 24;
  const long E = in_sizes[3];
  const int* src = eidx;
  const int* dst = eidx + E;

  // Byte-granular workspace allocator (256B aligned).
  char* base = (char*)d_ws;
  size_t off = 0;
  auto alloc = [&](size_t bytes) -> void* {
    void* p = base + off;
    off += (bytes + 255) & ~(size_t)255;
    return p;
  };
  float*    acc    = (float*)alloc((size_t)N * 64 * 4);
  float*    deg    = (float*)alloc((size_t)N * 4);
  _Float16* xin_h  = (_Float16*)alloc((size_t)N * 96 * 2);
  float*    t256   = (float*)alloc((size_t)N * 256 * 4);
  _Float16* t256_h = (_Float16*)alloc((size_t)N * 256 * 2);
  float*    h_a    = (float*)alloc((size_t)N * 128 * 4);
  float*    h_b    = (float*)alloc((size_t)N * 128 * 4);
  _Float16* h_ah   = (_Float16*)alloc((size_t)N * 128 * 2);
  _Float16* h_bh   = (_Float16*)alloc((size_t)N * 128 * 2);
  _Float16* ein_h  = (_Float16*)alloc((size_t)E * 32 * 2);
  // ee_h (f16 [E,128]) and ef_raw (f32 [E,64]) have identical byte size and
  // disjoint lifetimes: ef_raw is consumed by LN before the first ee GEMM.
  _Float16* ee_h   = (_Float16*)alloc((size_t)E * 128 * 2);
  float*    ef_raw = (float*)ee_h;
  _Float16* ef_h   = (_Float16*)alloc((size_t)E * 64 * 2);
  float*    xl     = (float*)alloc((size_t)N * 128 * 4);
  float*    xr     = (float*)alloc((size_t)N * 128 * 4);
  float*    logit  = (float*)alloc((size_t)E * 4 * 4);
  unsigned* mkey   = (unsigned*)alloc((size_t)N * 4 * 4);
  float*    ssum   = (float*)alloc((size_t)N * 4 * 4);
  float*    aggr   = (float*)alloc((size_t)N * 128 * 4);
  _Float16* cat_h  = (_Float16*)alloc((size_t)N * 384 * 2);
  _Float16* emb_h  = (_Float16*)alloc((size_t)N * 128 * 2);
  float*    himp   = (float*)alloc((size_t)N * 128 * 4);
  // Packed f16 weights
  _Float16* bp_nw1 = (_Float16*)alloc((size_t)96 * 256 * 2);
  _Float16* bp_nw2 = (_Float16*)alloc((size_t)256 * 128 * 2);
  _Float16* bp_ew  = (_Float16*)alloc((size_t)32 * 64 * 2);
  _Float16* bp_wl[3]; _Float16* bp_wr[3]; _Float16* bp_we[3];
  for (int l = 0; l < 3; ++l) {
    bp_wl[l] = (_Float16*)alloc((size_t)128 * 128 * 2);
    bp_wr[l] = (_Float16*)alloc((size_t)128 * 128 * 2);
    bp_we[l] = (_Float16*)alloc((size_t)64 * 128 * 2);
  }
  _Float16* bp_fw1 = (_Float16*)alloc((size_t)384 * 256 * 2);
  _Float16* bp_fw2 = (_Float16*)alloc((size_t)256 * 128 * 2);
  _Float16* bp_iw1 = (_Float16*)alloc((size_t)128 * 128 * 2);

  float* emb    = (float*)d_out;                     // [N,128]
  float* scores = (float*)d_out + (size_t)N * 128;   // [N]

  const int T = 256;
  const int WPB = T / 32;  // waves (strips) per block
#define GEMM_BLOCKS(M_, N_, NT_) cdiv((long)((M_) / 16) * ((N_) / (16 * (NT_))), WPB)
#define PACK(Bsrc, Bdst, N_, K_, K32_) \
  pack_b<<<cdiv((long)(K32_) * (N_), T), T, 0, stream>>>(Bsrc, Bdst, N_, K_, K32_)

  // --- Pack all weight matrices into WMMA-friendly f16 tiles ---
  PACK(n_w1, bp_nw1, 256, 88, 96);
  PACK(n_w2, bp_nw2, 128, 256, 256);
  PACK(e_w,  bp_ew,  64, 18, 32);
  for (int l = 0; l < 3; ++l) {
    PACK((const float*)d_in[20 + 6 * l + 0], bp_wl[l], 128, 128, 128);
    PACK((const float*)d_in[20 + 6 * l + 1], bp_wr[l], 128, 128, 128);
    PACK((const float*)d_in[20 + 6 * l + 2], bp_we[l], 128, 64, 64);
  }
  PACK(f_w1, bp_fw1, 256, 384, 384);
  PACK(f_w2, bp_fw2, 128, 256, 256);
  PACK(i_w1, bp_iw1, 128, 128, 128);

  // --- Position encoder + node input assembly ---
  fill_f32<<<cdiv((long)N * 64, T), T, 0, stream>>>(acc, 0.0f, (long)N * 64);
  fill_f32<<<cdiv((long)N, T), T, 0, stream>>>(deg, 0.0f, (long)N);
  pos_edge_kernel<<<cdiv(E, T), T, 0, stream>>>(pos, src, dst, p_w1, p_b1, p_g1, p_be1,
                                                p_w2, p_b2, acc, deg, E);
  build_xin<<<cdiv((long)N, T), T, 0, stream>>>(x, acc, deg, xin_h, N);

  // --- Node encoder MLP: [N,96] -> gelu -> [N,256] -> [N,128] ---
  gemm_wmma<1, 8><<<GEMM_BLOCKS(N, 256, 8), T, 0, stream>>>(
      xin_h, bp_nw1, n_b1, (float*)nullptr, t256_h, N, 256, 96);
  gemm_wmma<0, 8><<<GEMM_BLOCKS(N, 128, 8), T, 0, stream>>>(
      t256_h, bp_nw2, n_b2, h_a, h_ah, N, 128, 256);

  // --- Edge encoder: [E,32] -> [E,64] -> LN -> GELU (f16) ---
  build_ein<<<cdiv(E, T), T, 0, stream>>>(ea, et, ein_h, E);
  gemm_wmma<0, 4><<<GEMM_BLOCKS(E, 64, 4), T, 0, stream>>>(
      ein_h, bp_ew, e_b, ef_raw, (_Float16*)nullptr, (int)E, 64, 32);
  ln_gelu_rows_f16<<<cdiv(E, T), T, 0, stream>>>(ef_raw, ef_h, e_g, e_be, E, 64);

  // --- 3 GATv2 layers ---
  const int heads[3] = {4, 4, 1};
  const int dhs[3]   = {32, 32, 128};
  float*    h_cur  = h_a;  float*    h_nxt  = h_b;
  _Float16* h_curh = h_ah; _Float16* h_nxth = h_bh;
  for (int l = 0; l < 3; ++l) {
    const float* att  = (const float*)d_in[20 + 6 * l + 3];
    const float* ln_g = (const float*)d_in[20 + 6 * l + 4];
    const float* ln_b = (const float*)d_in[20 + 6 * l + 5];
    const int H = heads[l], dh = dhs[l];

    gemm_wmma<0, 8><<<GEMM_BLOCKS(N, 128, 8), T, 0, stream>>>(
        h_curh, bp_wl[l], (const float*)nullptr, xl, (_Float16*)nullptr, N, 128, 128);
    gemm_wmma<0, 8><<<GEMM_BLOCKS(N, 128, 8), T, 0, stream>>>(
        h_curh, bp_wr[l], (const float*)nullptr, xr, (_Float16*)nullptr, N, 128, 128);
    // Dominant streaming GEMM: grp_n == 1 -> ef_h read exactly once.
    gemm_wmma<0, 8><<<GEMM_BLOCKS(E, 128, 8), T, 0, stream>>>(
        ef_h, bp_we[l], (const float*)nullptr, (float*)nullptr, ee_h, (int)E, 128, 64);

    fill_u32<<<cdiv((long)N * H, T), T, 0, stream>>>(mkey, FKEY_NEG_INF, (long)N * H);
    fill_f32<<<cdiv((long)N * H, T), T, 0, stream>>>(ssum, 0.0f, (long)N * H);
    fill_f32<<<cdiv((long)N * 128, T), T, 0, stream>>>(aggr, 0.0f, (long)N * 128);

    att_logits_max<<<cdiv(E, T), T, 0, stream>>>(xl, xr, ee_h, src, dst, att,
                                                 logit, mkey, E, H, dh);
    att_expsum<<<cdiv(E, T), T, 0, stream>>>(logit, dst, mkey, ssum, E, H);
    att_scatter<<<cdiv(E * 4, T), T, 0, stream>>>(xl, logit, src, dst, mkey, ssum,
                                                  aggr, E, H, dh);
    residual_ln_gelu<<<cdiv((long)N, T), T, 0, stream>>>(
        aggr, h_cur, ln_g, ln_b, h_nxt, h_nxth, cat_h, l * 128, N);

    float* tf = h_cur; h_cur = h_nxt; h_nxt = tf;
    _Float16* th = h_curh; h_curh = h_nxth; h_nxth = th;
  }

  // --- Fusion: [N,384] -> [N,256] -> LN -> GELU -> [N,128] = emb ---
  gemm_wmma<0, 8><<<GEMM_BLOCKS(N, 256, 8), T, 0, stream>>>(
      cat_h, bp_fw1, f_b1, t256, (_Float16*)nullptr, N, 256, 384);
  ln_gelu_rows_f16<<<cdiv((long)N, T), T, 0, stream>>>(t256, t256_h, f_g, f_be, (long)N, 256);
  gemm_wmma<0, 8><<<GEMM_BLOCKS(N, 128, 8), T, 0, stream>>>(
      t256_h, bp_fw2, f_b2, emb, emb_h, N, 128, 256);

  // --- Importance head: gelu(emb @ w1 + b1) @ w2 + b2 ---
  gemm_wmma<1, 8><<<GEMM_BLOCKS(N, 128, 8), T, 0, stream>>>(
      emb_h, bp_iw1, i_b1, himp, (_Float16*)nullptr, N, 128, 128);
  imp_final<<<cdiv((long)N, T), T, 0, stream>>>(himp, i_w2, i_b2, scores, N);
#undef GEMM_BLOCKS
#undef PACK
}
